// MHADisSubFormer_44452911514086
// MI455X (gfx1250) — matrix operand order repaired and checked by reading, not compiled, not run on hardware
//
#include <hip/hip_runtime.h>
#include <hip/hip_bf16.h>

// MHADisSubFormer fused kernel for gfx1250 (MI455X).
// B=16, N=256, A=64, D=256.
//
// Algebra: score_a = (q@Wk)·x_a + q·bk ; out = Wv·(Σ p_a x_a) + (Σ p_a)·bv
// -> anchors (256 MB) are streamed exactly ONCE in the common case (online
//    softmax); all projections are 16x256x256 GEMMs on v_wmma_f32_16x16x32_bf16.
//    Memory-bound: ~265 MB @ 23.3 TB/s ≈ 11.5 us floor.

#define BB 16
#define NN 256
#define AA 64
#define DD 256
#define NW 16          // waves per block (wave32)
#define WSZ 32

typedef __bf16 bf16_t;
typedef __attribute__((ext_vector_type(16))) __bf16 v16bf;
typedef __attribute__((ext_vector_type(8)))  float  v8f;

// A-matrix (16x32 bf16) per-lane K offset: lanes 0-15 hold K {0..7,16..23},
// lanes 16-31 hold K {8..15,24..31} (ISA 7.12.2).
__device__ __forceinline__ int koffA(int lane, int e) {
  int kb = (lane < 16) ? 0 : 8;
  return (e < 8) ? (kb + e) : (kb + 16 + (e - 8));
}
// B-matrix (32x16 bf16): same K order as A (lanes 0-15 K {0..7,16..23}, lanes 16-31 K {8..15,24..31}).
__device__ __forceinline__ int koffB(int lane, int e) {
  return koffA(lane, e);
}

__device__ __forceinline__ float waveRedAdd(float v) {
#pragma unroll
  for (int m = 16; m >= 1; m >>= 1) v += __shfl_xor(v, m, 32);
  return v;
}

__global__ __launch_bounds__(NW * WSZ)
void mha_dissub_kernel(const float* __restrict__ cc,       // [B,N,D]
                       const float* __restrict__ anchors,  // [B,N,A,D]
                       const float* __restrict__ sims,     // [B,N,A]
                       const float* __restrict__ Wq, const float* __restrict__ bq,
                       const float* __restrict__ Wk, const float* __restrict__ bk,
                       const float* __restrict__ Wv, const float* __restrict__ bv,
                       float* __restrict__ out)            // [B,N,1,D]
{
  __shared__ bf16_t ccB[16][DD];    // cc rows, bf16              (8 KB)
  __shared__ bf16_t QA[16][DD];     // Q = cc@Wq.T+bq, bf16       (8 KB)
  __shared__ float  Qp[16][DD];     // (Q@Wk)/sqrt(D), f32        (16 KB)
  __shared__ bf16_t Y[16][DD];      // weighted anchor sums, bf16 (8 KB)
  __shared__ float  simsL[16][AA];  // sims tile                  (4 KB)
  __shared__ float  bqL[DD], bkL[DD], bvL[DD];
  __shared__ float  sumP[16];

  const int tid  = threadIdx.x;
  const int wave = tid >> 5;
  const int lane = tid & 31;
  const int b    = blockIdx.x >> 4;        // B=16 -> grid = B * N/16 = 256
  const int n0   = (blockIdx.x & 15) << 4; // 16 tokens per block

  // ---- Phase 0: stage cc (as bf16), sims, biases into LDS ----
  {
    const size_t ccbase = (size_t)(b * NN + n0) * DD;
#pragma unroll 2
    for (int idx = tid; idx < 16 * DD; idx += NW * WSZ)
      (&ccB[0][0])[idx] = (bf16_t)cc[ccbase + idx];
    const size_t sbase = (size_t)(b * NN + n0) * AA;
    for (int idx = tid; idx < 16 * AA; idx += NW * WSZ)
      (&simsL[0][0])[idx] = sims[sbase + idx];
    if (tid < DD) { bqL[tid] = bq[tid]; bkL[tid] = bk[tid]; bvL[tid] = bv[tid]; }
  }
  __syncthreads();

  const int col    = lane & 15;
  const int rowoff = (lane < 16) ? 0 : 8;

  // ---- Phase 1: Q16 = cc16x256 @ Wq.T + bq  (WMMA, bf16 in / f32 acc) ----
  {
    const int d0 = wave << 4;                 // each wave: one 16-col tile
    v8f acc = {};
#pragma unroll
    for (int kb = 0; kb < DD; kb += 32) {
      v16bf a, bm;
      const float* wrow = Wq + (size_t)(d0 + col) * DD + kb; // B[k,c]=Wq[d0+c, kb+k]
#pragma unroll
      for (int e = 0; e < 16; ++e) {
        a[e]  = ccB[col][kb + koffA(lane, e)];
        bm[e] = (bf16_t)wrow[koffB(lane, e)];
      }
      acc = __builtin_amdgcn_wmma_f32_16x16x32_bf16(false, a, false, bm,
                                                    (short)0, acc, false, false);
    }
#pragma unroll
    for (int v = 0; v < 8; ++v) {
      int r = v + rowoff;
      QA[r][d0 + col] = (bf16_t)(acc[v] + bqL[d0 + col]);
    }
  }
  __syncthreads();

  // ---- Phase 2: Qp = (Q16 @ Wk) * (1/sqrt(D))  (WMMA) ----
  {
    const int e0 = wave << 4;
    v8f acc = {};
#pragma unroll
    for (int kb = 0; kb < DD; kb += 32) {
      v16bf a, bm;
#pragma unroll
      for (int e = 0; e < 16; ++e) {
        a[e]  = QA[col][kb + koffA(lane, e)];
        bm[e] = (bf16_t)Wk[(size_t)(kb + koffB(lane, e)) * DD + e0 + col]; // B[k,c]=Wk[k, e0+c]
      }
      acc = __builtin_amdgcn_wmma_f32_16x16x32_bf16(false, a, false, bm,
                                                    (short)0, acc, false, false);
    }
#pragma unroll
    for (int v = 0; v < 8; ++v)
      Qp[v + rowoff][e0 + col] = acc[v] * 0.0625f; // 1/sqrt(256)
  }
  __syncthreads();

  // ---- Phase 3: one wave per token; single streaming pass over anchors
  //      with online softmax (flash style), software-pipelined loads ----
  {
    const int n = n0 + wave;
    float qv[8];
#pragma unroll
    for (int i = 0; i < 8; ++i) qv[i] = Qp[wave][lane * 8 + i];

    // c_n = (Q[n]·bk)/sqrt(D)
    float part = 0.f;
#pragma unroll
    for (int i = 0; i < 8; ++i)
      part += (float)QA[wave][lane * 8 + i] * bkL[lane * 8 + i];
    const float c_n = waveRedAdd(part) * 0.0625f;

    const float4* abase =
        (const float4*)(anchors + (size_t)(b * NN + n) * AA * DD);
    const int lx = lane * 2;

    float m = -1e30f, l = 0.f, rawsum = 0.f;
    float acc[8];
#pragma unroll
    for (int i = 0; i < 8; ++i) acc[i] = 0.f;

    // software pipeline: current row in (c0,c1), next row prefetched
    float4 c0 = abase[lx];
    float4 c1 = abase[lx + 1];
    for (int a = 0; a < AA; ++a) {
      float4 p0, p1;
      if (a + 1 < AA) {
        p0 = abase[(a + 1) * 64 + lx];          // global_load_b128 x2 (next row)
        p1 = abase[(a + 1) * 64 + lx + 1];
      }
      if (a + 4 < AA)                            // global_prefetch_b8, 4 rows ahead
        __builtin_prefetch(&abase[(a + 4) * 64 + lx], 0, 1);

      float x[8] = {c0.x, c0.y, c0.z, c0.w, c1.x, c1.y, c1.z, c1.w};
      float p = 0.f;
#pragma unroll
      for (int i = 0; i < 8; ++i) p += x[i] * qv[i];
      const float s = waveRedAdd(p) + c_n + simsL[wave][a];
      rawsum += s;

      const float mn   = fmaxf(m, s);
      const float corr = __expf(m - mn);
      const float w    = __expf(s - mn);
      l = l * corr + w;
#pragma unroll
      for (int i = 0; i < 8; ++i) acc[i] = acc[i] * corr + w * x[i];
      m = mn;

      c0 = p0; c1 = p1;
    }

    float y[8];
    float sp;
    if (rawsum != 0.f) {                 // wave-uniform branch (common path)
      sp = 1.f;
      const float inv_l = 1.f / l;
#pragma unroll
      for (int i = 0; i < 8; ++i) y[i] = acc[i] * inv_l;
    } else {
      // Rare passthrough path (Σadj == 0): probs = adj, Σp = 0.
      // Re-stream this token's anchors (64 KB) to build Σ s_a * x_a;
      // s_a recomputed identically to the first pass.
      sp = 0.f;
#pragma unroll
      for (int i = 0; i < 8; ++i) y[i] = 0.f;
      for (int a = 0; a < AA; ++a) {
        const float4 x0 = abase[a * 64 + lx];
        const float4 x1 = abase[a * 64 + lx + 1];
        float x[8] = {x0.x, x0.y, x0.z, x0.w, x1.x, x1.y, x1.z, x1.w};
        float p = 0.f;
#pragma unroll
        for (int i = 0; i < 8; ++i) p += x[i] * qv[i];
        const float s = waveRedAdd(p) + c_n + simsL[wave][a];
#pragma unroll
        for (int i = 0; i < 8; ++i) y[i] += s * x[i];
      }
    }
#pragma unroll
    for (int i = 0; i < 8; ++i) Y[wave][lane * 8 + i] = (bf16_t)y[i];
    if (lane == 0) sumP[wave] = sp;
  }
  __syncthreads();

  // ---- Phase 4: Out16 = Y16 @ Wv.T + sumP*bv  (WMMA), write f32 ----
  {
    const int d0 = wave << 4;
    v8f acc = {};
#pragma unroll
    for (int kb = 0; kb < DD; kb += 32) {
      v16bf a, bm;
      const float* wrow = Wv + (size_t)(d0 + col) * DD + kb;
#pragma unroll
      for (int e = 0; e < 16; ++e) {
        a[e]  = Y[col][kb + koffA(lane, e)];
        bm[e] = (bf16_t)wrow[koffB(lane, e)];
      }
      acc = __builtin_amdgcn_wmma_f32_16x16x32_bf16(false, a, false, bm,
                                                    (short)0, acc, false, false);
    }
#pragma unroll
    for (int v = 0; v < 8; ++v) {
      int r = v + rowoff;
      out[(size_t)(b * NN + n0 + r) * DD + d0 + col] =
          acc[v] + sumP[r] * bvL[d0 + col];
    }
  }
}

extern "C" void kernel_launch(void* const* d_in, const int* in_sizes, int n_in,
                              void* d_out, int out_size, void* d_ws, size_t ws_size,
                              hipStream_t stream) {
  const float* cc      = (const float*)d_in[0];
  const float* anchors = (const float*)d_in[1];
  const float* sims    = (const float*)d_in[2];
  const float* Wq      = (const float*)d_in[3];
  const float* bq      = (const float*)d_in[4];
  const float* Wk      = (const float*)d_in[5];
  const float* bk      = (const float*)d_in[6];
  const float* Wv      = (const float*)d_in[7];
  const float* bv      = (const float*)d_in[8];
  float* out = (float*)d_out;

  dim3 grid(BB * (NN / 16));   // 256 blocks
  dim3 block(NW * WSZ);        // 512 threads = 16 wave32
  mha_dissub_kernel<<<grid, block, 0, stream>>>(cc, anchors, sims,
                                                Wq, bq, Wk, bk, Wv, bv, out);
}
